// PooledSSIMLoss_79645873537527
// MI455X (gfx1250) — matrix operand order repaired and checked by reading, not compiled, not run on hardware
//
#include <hip/hip_runtime.h>
#include <stdint.h>

// ---- CDNA5 vector types ----
typedef _Float16 v16h __attribute__((ext_vector_type(16)));
typedef float    v8f  __attribute__((ext_vector_type(8)));
typedef uint32_t v4u  __attribute__((ext_vector_type(4)));
typedef int      v8i  __attribute__((ext_vector_type(8)));
typedef int      v4i  __attribute__((ext_vector_type(4)));

#define H_IN    512
#define W_IN    512
#define PLANES  48          // 16 batch * 3 channels
#define OH      502
#define OW      502
#define OH_T    16          // output rows per tile
#define OW_T    64          // output cols per tile
#define TROWS   32          // LDS tile rows (padded to 2 WMMA rowblocks)
#define TDROWS  26          // rows actually DMA'd (16 + 10 halo)
#define TCOLS   84          // LDS tile cols / pitch (64 + 10 halo, padded)
#define HROWS   26          // valid horizontally-summed rows
#define HROWSP  32          // padded H rows -> unconditional WMMA-D stores
#define NBLK    (8 * 32 * 48)

// Issue one TDM DMA: 2D tile (TDROWS x TCOLS fp32) from global -> LDS.
// tensor_dim0/1 are the remaining extent from the tile origin so OOB
// elements zero-fill (no manual clamping, no OOB reads).
__device__ __forceinline__ void tdm_load_tile(const float* gptr,
                                              uint32_t lds_byte_addr,
                                              int cols_rem, int rows_rem) {
  uint64_t ga = (uint64_t)(uintptr_t)gptr;
  v4u g0;
  g0.x = 1u;                                            // count=1 (valid descriptor)
  g0.y = lds_byte_addr;                                 // D#.lds_addr
  g0.z = (uint32_t)(ga & 0xFFFFFFFFu);                  // global_addr[31:0]
  g0.w = (uint32_t)((ga >> 32) & 0x01FFFFFFu)           // global_addr[56:32]
       | (2u << 30);                                    // type=2 ("image")
  v8i g1;
  g1[0] = (int)(2u << 16);                              // workgroup_mask=0, data_size=2 (4B)
  g1[1] = (int)(((uint32_t)cols_rem & 0xFFFFu) << 16);  // abar=0 | tensor_dim0[15:0]
  g1[2] = (int)((((uint32_t)cols_rem >> 16) & 0xFFFFu)  // tensor_dim0[31:16]
       | (((uint32_t)rows_rem & 0xFFFFu) << 16));       // tensor_dim1[15:0]
  g1[3] = (int)((((uint32_t)rows_rem >> 16) & 0xFFFFu)  // tensor_dim1[31:16]
       | ((uint32_t)TCOLS << 16));                      // tile_dim0 (= LDS pitch)
  g1[4] = (int)TDROWS;                                  // tile_dim1 (tile_dim2=0 -> 2D)
  g1[5] = (int)W_IN;                                    // tensor_dim0_stride[31:0]
  g1[6] = 0;                                            // stride hi / dim1_stride lo (unused 2D)
  g1[7] = 0;
  v4i gz4 = {0, 0, 0, 0};                               // groups 2/3 unused (<=2D)
  v8i gz8 = {0, 0, 0, 0, 0, 0, 0, 0};                   // extra group (6-arg toolchain form)
  __builtin_amdgcn_tensor_load_to_lds(g0, g1, gz4, gz4, gz8, 0);
}

__global__ __launch_bounds__(256)
void ssim_main_kernel(const float* __restrict__ x,
                      const float* __restrict__ y,
                      float* __restrict__ part) {
  __shared__ float xtile[TROWS * TCOLS];                // rows >= TDROWS uninit (discarded)
  __shared__ float ytile[TROWS * TCOLS];
  __shared__ float Hs[5][HROWSP * OW_T];                // rows >= HROWS are write-only pad

  const int tid   = threadIdx.x;
  const int c0    = blockIdx.x * OW_T;
  const int r0    = blockIdx.y * OH_T;
  const int plane = blockIdx.z;

  const size_t pbase = (size_t)plane * (H_IN * W_IN) + (size_t)r0 * W_IN + c0;

  // ---- Stage 1: TDM global -> LDS (wave 0 issues, tracked by TENSORcnt) ----
  if (tid < 32) {
    tdm_load_tile(x + pbase, (uint32_t)(uintptr_t)xtile, W_IN - c0, H_IN - r0);
    tdm_load_tile(y + pbase, (uint32_t)(uintptr_t)ytile, W_IN - c0, H_IN - r0);
    __builtin_amdgcn_s_wait_tensorcnt(0);
  }
  __syncthreads();

  // ---- Stage 2: horizontal 11-sums via WMMA banded matmul ----
  // Out16x16 = A(16x26 data, f16 hi/lo split) x Band(26x16 ones), K padded to 32.
  // D row m depends only on A row m, so garbage A rows (>= TDROWS) only
  // produce garbage D rows (>= HROWS), which land in the write-only pad.
  const int wave = tid >> 5;            // 0..7
  const int lane = tid & 31;
  const int rb   = (wave >> 2) * 16;    // rowblock base: 0 or 16
  const int cb   = (wave & 3) * 16;     // colblock base: 0,16,32,48
  const int mrow = lane & 15;
  const int hiH  = lane >> 4;           // 0 or 1

  // A-matrix source values (ISA 16-bit A layout: lanes0-15 h0-7->K0-7,h8-15->K16-23;
  // lanes16-31 h0-7->K8-15, h8-15->K24-31)
  float xv[16], yv[16];
#pragma unroll
  for (int h = 0; h < 16; ++h) {
    const int K   = h + ((h & 8) ? 8 : 0) + (hiH ? 8 : 0);
    const int idx = (rb + mrow) * TCOLS + (cb + K);
    xv[h] = xtile[idx];
    yv[h] = ytile[idx];
  }

  // B band matrix (same K order as A: lanes0-15 h0-7->K0-7,h8-15->K16-23; lanes16-31 h0-7->K8-15,h8-15->K24-31)
  v16h bfrag;
#pragma unroll
  for (int h = 0; h < 16; ++h) {
    const int K = h + ((h & 8) ? 8 : 0) + (hiH ? 8 : 0);
    const int d = K - mrow;
    bfrag[h] = (d >= 0 && d <= 10) ? (_Float16)1.0f : (_Float16)0.0f;
  }

#pragma unroll
  for (int q = 0; q < 5; ++q) {
    v16h ahi, alo;
#pragma unroll
    for (int h = 0; h < 16; ++h) {
      float v = (q == 0) ? xv[h]
              : (q == 1) ? yv[h]
              : (q == 2) ? xv[h] * xv[h]
              : (q == 3) ? yv[h] * yv[h]
              :            xv[h] * yv[h];
      _Float16 hi = (_Float16)v;
      ahi[h] = hi;
      alo[h] = (_Float16)(v - (float)hi);   // residual: linear op => exact recombine
    }
    v8f acc = {0.f, 0.f, 0.f, 0.f, 0.f, 0.f, 0.f, 0.f};
    acc = __builtin_amdgcn_wmma_f32_16x16x32_f16(false, ahi, false, bfrag,
                                                 (short)0, acc, false, false);
    acc = __builtin_amdgcn_wmma_f32_16x16x32_f16(false, alo, false, bfrag,
                                                 (short)0, acc, false, false);
    // D layout: VGPR i -> row rb + i + (lane>=16 ? 8 : 0), col cb + (lane&15)
    // Unconditional stores: padded Hs absorbs rows 26..31.
#pragma unroll
    for (int i = 0; i < 8; ++i) {
      const int orow = rb + i + (hiH ? 8 : 0);
      Hs[q][orow * OW_T + (cb + mrow)] = acc[i];
    }
  }
  __syncthreads();

  // ---- Stage 3: vertical 11-sum sliding window + SSIM map ----
  const int cc  = tid & 63;
  const int rs  = tid >> 6;   // 0..3, 4 rows each
  const int rb3 = rs * 4;

  float s0 = 0.f, s1 = 0.f, s2 = 0.f, s3 = 0.f, s4 = 0.f;
#pragma unroll
  for (int k = 0; k < 10; ++k) {
    const int o = (rb3 + k) * OW_T + cc;
    s0 += Hs[0][o]; s1 += Hs[1][o]; s2 += Hs[2][o]; s3 += Hs[3][o]; s4 += Hs[4][o];
  }

  const float C1 = 1e-4f, C2 = 9e-4f, INV = 1.0f / 121.0f, EPS = 1e-8f;
  float acc = 0.f;
#pragma unroll
  for (int rr = 0; rr < 4; ++rr) {
    const int r  = rb3 + rr;
    const int oT = (r + 10) * OW_T + cc;       // max row index = 25 < HROWS
    s0 += Hs[0][oT]; s1 += Hs[1][oT]; s2 += Hs[2][oT]; s3 += Hs[3][oT]; s4 += Hs[4][oT];

    const float mx  = s0 * INV, my = s1 * INV;
    const float ex2 = s2 * INV, ey2 = s3 * INV, exy = s4 * INV;
    const float mx2 = mx * mx, my2 = my * my, mxy = mx * my;
    const float sx  = fmaxf(ex2 - mx2, 0.f);
    const float sy  = fmaxf(ey2 - my2, 0.f);
    const float sxy = exy - mxy;
    const float num = (2.f * mxy + C1) * (2.f * sxy + C2);
    const float den = (mx2 + my2 + C1) * (sx + sy + C2) + EPS;
    if ((r0 + r) < OH && (c0 + cc) < OW) acc += num / den;

    const int oB = r * OW_T + cc;
    s0 -= Hs[0][oB]; s1 -= Hs[1][oB]; s2 -= Hs[2][oB]; s3 -= Hs[3][oB]; s4 -= Hs[4][oB];
  }

  // ---- Deterministic block reduction (reuse xtile as scratch) ----
  float* red = xtile;
  red[tid] = acc;
  __syncthreads();
#pragma unroll
  for (int s = 128; s > 0; s >>= 1) {
    if (tid < s) red[tid] += red[tid + s];
    __syncthreads();
  }
  if (tid == 0) {
    const int bid = (blockIdx.z * gridDim.y + blockIdx.y) * gridDim.x + blockIdx.x;
    part[bid] = red[0];
  }
}

__global__ __launch_bounds__(256)
void ssim_reduce_kernel(const float* __restrict__ part, float* __restrict__ out) {
  __shared__ float red[256];
  float a = 0.f;
  for (int i = threadIdx.x; i < NBLK; i += 256) a += part[i];
  red[threadIdx.x] = a;
  __syncthreads();
#pragma unroll
  for (int s = 128; s > 0; s >>= 1) {
    if (threadIdx.x < s) red[threadIdx.x] += red[threadIdx.x + s];
    __syncthreads();
  }
  if (threadIdx.x == 0) {
    const float count = (float)PLANES * (float)OH * (float)OW;  // 12,096,192
    out[0] = 1.0f - red[0] / count;
  }
}

extern "C" void kernel_launch(void* const* d_in, const int* in_sizes, int n_in,
                              void* d_out, int out_size, void* d_ws, size_t ws_size,
                              hipStream_t stream) {
  const float* x = (const float*)d_in[0];
  const float* y = (const float*)d_in[1];
  float* part = (float*)d_ws;   // NBLK floats, fully overwritten each call
  float* out  = (float*)d_out;

  dim3 grid(8, 32, PLANES);     // col-tiles, row-tiles, planes
  ssim_main_kernel<<<grid, 256, 0, stream>>>(x, y, part);
  ssim_reduce_kernel<<<1, 256, 0, stream>>>(part, out);
}
